// Where2comm_55293408969327
// MI455X (gfx1250) — matrix operand order repaired and checked by reading, not compiled, not run on hardware
//
#include <hip/hip_runtime.h>
#include <hip/hip_bf16.h>
#include <math.h>

#define Bv 4
#define Lv 5
#define Cv 256
#define Hv 96
#define Wv 192
#define HWv (Hv*Wv)
#define BLHW (Bv*Lv*HWv)
#define KS 5
#define THRE_C 0.01f

typedef __attribute__((ext_vector_type(16))) __bf16 v16bf;
typedef __attribute__((ext_vector_type(8)))  __bf16 v8bf;
typedef __attribute__((ext_vector_type(8)))  float  v8f;

__device__ __forceinline__ float sigm(float x){ return 1.0f/(1.0f + __expf(-x)); }

__device__ __forceinline__ float lane_bcast(float v, int l){
  return __builtin_bit_cast(float, __builtin_amdgcn_readlane(__builtin_bit_cast(int, v), l));
}

// Workspace layout (floats): [0] comm accumulator, conf at +64, mask at +64+BLHW.

__global__ void k_conf(const float* __restrict__ rm, float* __restrict__ ws){
  int i = blockIdx.x*blockDim.x + threadIdx.x;
  if (i == 0) ws[0] = 0.0f;                     // zero comm accumulator
  if (i >= BLHW) return;
  int bl = i / HWv, pos = i % HWv;
  const float* r = rm + (size_t)bl*2*HWv + pos; // ANCHOR=2
  ws[64 + i] = fmaxf(sigm(r[0]), sigm(r[HWv]));
}

__global__ void k_mask(const float* __restrict__ g, float* __restrict__ ws){
  int i = blockIdx.x*blockDim.x + threadIdx.x;
  if (i >= BLHW) return;
  int bl = i / HWv, pos = i % HWv;
  int h = pos / Wv, w = pos % Wv;
  const float* conf = ws + 64 + (size_t)bl*HWv;
  float s = 0.0f;
  #pragma unroll
  for (int ky = 0; ky < KS; ++ky){
    int y = h + ky - 2;
    if (y < 0 || y >= Hv) continue;
    #pragma unroll
    for (int kx = 0; kx < KS; ++kx){
      int x = w + kx - 2;
      if (x < 0 || x >= Wv) continue;
      s += g[ky*KS + kx] * conf[y*Wv + x];
    }
  }
  float m = (s > THRE_C) ? 1.0f : 0.0f;
  if ((bl % Lv) == 0){
    if (m != 0.0f) atomicAdd(ws, 1.0f);         // exact integer sums -> deterministic
    m = 1.0f;                                   // mask[:,0] forced to 1 after counting
  }
  ws[64 + BLHW + i] = m;
}

__global__ __launch_bounds__(256) void k_fuse(const float* __restrict__ x,
                                              const float* __restrict__ tm,
                                              const float* __restrict__ ws,
                                              float* __restrict__ out){
  __shared__ float featF[8][Lv][Cv];                    // f32 copy for the combine
  __shared__ __align__(16) __bf16 featH[8][Lv][272];    // bf16 copy for WMMA (16B-aligned rows)

  const int tid  = threadIdx.x;
  const int wave = tid >> 5, lane = tid & 31;
  const int pix  = blockIdx.x*8 + wave;                 // grid divides exactly -> no divergence
  const int b    = pix / HWv;
  const int pos  = pix % HWv;
  const int h    = pos / Wv, w = pos % Wv;
  const float* mask = ws + 64 + BLHW;

  const float gx = (w + 0.5f)*(2.0f/Wv) - 1.0f;
  const float gy = (h + 0.5f)*(2.0f/Hv) - 1.0f;
  const float s01 = (float)Hv/(float)Wv;
  const float s02 = 2.0f/(0.4f*(float)Wv);              // DOWNSAMPLE*DISCRETE_RATIO = 0.4
  const float s10 = (float)Wv/(float)Hv;
  const float s12 = 2.0f/(0.4f*(float)Hv);

  for (int l = 0; l < Lv; ++l){
    const float* M = tm + ((size_t)(b*Lv + 0)*Lv + l)*16;   // theta = t2[:,0]
    float t00 = M[0],       t01 = M[1]*s01, t02 = M[3]*s02;
    float t10 = M[4]*s10,   t11 = M[5],     t12 = M[7]*s12;
    float u = t00*gx + t01*gy + t02;
    float v = t10*gx + t11*gy + t12;
    float sx = ((u + 1.0f)*Wv - 1.0f)*0.5f;
    float sy = ((v + 1.0f)*Hv - 1.0f)*0.5f;
    float fx0 = floorf(sx), fy0 = floorf(sy);
    float wx = sx - fx0, wy = sy - fy0;
    int ix0 = (int)fx0, iy0 = (int)fy0;
    float tw[4] = {(1.0f-wx)*(1.0f-wy), wx*(1.0f-wy), (1.0f-wx)*wy, wx*wy};
    const int dxs[4] = {0,1,0,1}, dys[4] = {0,0,1,1};
    const int bl = b*Lv + l;
    const float* mp = mask + (size_t)bl*HWv;
    float mw[4]; size_t off[4];
    #pragma unroll
    for (int t = 0; t < 4; ++t){
      int ix = ix0 + dxs[t], iy = iy0 + dys[t];
      bool valid = (ix >= 0) & (ix < Wv) & (iy >= 0) & (iy < Hv);
      int ixc = ix < 0 ? 0 : (ix > Wv-1 ? Wv-1 : ix);
      int iyc = iy < 0 ? 0 : (iy > Hv-1 ? Hv-1 : iy);
      float mv = valid ? mp[iyc*Wv + ixc] : 0.0f;        // x*mask sampled at source tap
      mw[t]  = tw[t] * mv;
      off[t] = ((size_t)bl*Cv)*HWv + (size_t)iyc*Wv + ixc;
    }
    #pragma unroll
    for (int j = 0; j < 8; ++j){
      int c = lane + 32*j;
      size_t coff = (size_t)c*HWv;
      float acc = mw[0]*x[off[0]+coff] + mw[1]*x[off[1]+coff]
                + mw[2]*x[off[2]+coff] + mw[3]*x[off[3]+coff];
      featF[wave][l][c] = acc;
      featH[wave][l][c] = (__bf16)acc;
    }
  }
  __syncthreads();   // publish LDS feat tiles (cross-lane reads below)

  // --- 5x5 gram via v_wmma_f32_16x16x32_bf16, M/N padded to 16, K = 256 (8 chunks) ---
  int m = lane & 15; if (m > Lv-1) m = Lv-1;             // clamp padding rows (harmless dup)
  const __bf16* rowp = &featH[wave][m][0];
  const v8bf* r8 = (const v8bf*)rowp;
  const int khA = (lane >> 4) * 8;                       // A striping: {0-7,16-23}/{8-15,24-31}
  const int khB = (lane >> 4) * 8;                       // B striping: {0-7,16-23}/{8-15,24-31}
  v8f acc = {};
  #pragma unroll
  for (int kk = 0; kk < 8; ++kk){
    const int kb = kk*32;
    v8bf a0 = r8[(kb + khA) >> 3];
    v8bf a1 = r8[(kb + 16 + khA) >> 3];
    v8bf b0 = r8[(kb + khB) >> 3];
    v8bf b1 = r8[(kb + 16 + khB) >> 3];
    v16bf A, Bf;
    #pragma unroll
    for (int i = 0; i < 8; ++i){ A[i] = a0[i]; A[8+i] = a1[i]; Bf[i] = b0[i]; Bf[8+i] = b1[i]; }
    acc = __builtin_amdgcn_wmma_f32_16x16x32_bf16(false, A, false, Bf, (short)0, acc, false, false);
  }

  // D row M=0 lives in acc[0], lanes 0..4 hold score[0, n]
  float sc[Lv];
  #pragma unroll
  for (int n = 0; n < Lv; ++n) sc[n] = lane_bcast(acc[0], n) * (1.0f/16.0f);  // 1/sqrt(256)
  float mx = sc[0];
  #pragma unroll
  for (int n = 1; n < Lv; ++n) mx = fmaxf(mx, sc[n]);
  float p[Lv], sum = 0.0f;
  #pragma unroll
  for (int n = 0; n < Lv; ++n){ p[n] = __expf(sc[n] - mx); sum += p[n]; }
  float inv = 1.0f/sum;

  #pragma unroll
  for (int j = 0; j < 8; ++j){
    int c = lane + 32*j;
    float o = 0.0f;
    #pragma unroll
    for (int n = 0; n < Lv; ++n) o += (p[n]*inv) * featF[wave][n][c];
    out[(((size_t)b*Cv + c)*Hv + h)*Wv + w] = o;
  }
}

__global__ void k_comm(const float* __restrict__ ws, float* __restrict__ out){
  out[0] = ws[0] / (float)(Bv*HWv);   // mean over B, normalized by H*W
}

extern "C" void kernel_launch(void* const* d_in, const int* in_sizes, int n_in,
                              void* d_out, int out_size, void* d_ws, size_t ws_size,
                              hipStream_t stream){
  const float* x  = (const float*)d_in[0];   // (B*L, C, H, W)
  const float* rm = (const float*)d_in[1];   // (B*L, 2, H, W)
  const float* tm = (const float*)d_in[2];   // (B, L, L, 4, 4)
  const float* g  = (const float*)d_in[3];   // (5, 5)
  // d_in[4] = cav_num (unused; L fixed)
  float* ws  = (float*)d_ws;
  float* out = (float*)d_out;

  k_conf<<<(BLHW + 255)/256, 256, 0, stream>>>(rm, ws);
  k_mask<<<(BLHW + 255)/256, 256, 0, stream>>>(g, ws);
  k_fuse<<<(Bv*HWv)/8, 256, 0, stream>>>(x, tm, ws, out);
  k_comm<<<1, 1, 0, stream>>>(ws, out + (size_t)Bv*Cv*HWv);
}
